// QwenAttention_65343632441681
// MI455X (gfx1250) — matrix operand order repaired and checked by reading, not compiled, not run on hardware
//
#include <hip/hip_runtime.h>
#include <hip/hip_bf16.h>

// ---------------- problem constants ----------------
#define BB   8
#define HH   16
#define KVH  2
#define DD   128
#define SS   16
#define TT   8192
#define HID  2048
#define GG   8          // HH / KVH
#define NSPLIT 16
#define TCHUNK 64
#define XC   512        // x staging chunk (columns of HID)

typedef __attribute__((ext_vector_type(16))) __bf16       v16bf;
typedef __attribute__((ext_vector_type(8)))  float        v8f;
typedef __attribute__((ext_vector_type(4)))  float        f32x4;
typedef __attribute__((ext_vector_type(4)))  unsigned int u32x4;
typedef __attribute__((ext_vector_type(2)))  unsigned int u32x2;

union V16 {
  v16bf  bf;
  __bf16 h[16];
  u32x4  q[2];
};

// Load 32-wide K-slice of an A-matrix row (fp32 source, contiguous k=0..31 at p).
// k order used here (B matches): lane half=0 holds k={0..15}, half=1 holds k={16..31}.
__device__ __forceinline__ void loadA_f32(V16& a, const float* __restrict__ p, int half) {
  f32x4 r0 = *(const f32x4*)(p + 16 * half);
  f32x4 r1 = *(const f32x4*)(p + 16 * half + 4);
  f32x4 r2 = *(const f32x4*)(p + 16 * half + 8);
  f32x4 r3 = *(const f32x4*)(p + 16 * half + 12);
#pragma unroll
  for (int e = 0; e < 4; ++e) {
    a.h[e]      = (__bf16)r0[e];
    a.h[4 + e]  = (__bf16)r1[e];
    a.h[8 + e]  = (__bf16)r2[e];
    a.h[12 + e] = (__bf16)r3[e];
  }
}

__device__ __forceinline__ v8f wmma_bf16(const V16& a, const V16& b, v8f c) {
  return __builtin_amdgcn_wmma_f32_16x16x32_bf16(false, a.bf, false, b.bf,
                                                 (short)0, c, false, false);
}

// =====================================================================
// Kernel 1: fused QKV projection + RMSNorm + RoPE + KV-cache row update
// grid (20, B): slot 0..15 = q head, 16..17 = k head, 18..19 = v head
// q is written to workspace in bf16 (attention A-operand needs bf16 anyway).
// =====================================================================
__global__ __launch_bounds__(256) void qkv_kernel(
    const float* __restrict__ x,   const float* __restrict__ wq,
    const float* __restrict__ wk,  const float* __restrict__ wv,
    const float* __restrict__ cosb,const float* __restrict__ sinb,
    const float* __restrict__ qnw, const float* __restrict__ knw,
    const int*   __restrict__ posp,
    __bf16* __restrict__ q_rot, float* __restrict__ new_k, float* __restrict__ new_v)
{
  const int slot = blockIdx.x, b = blockIdx.y;
  const int tid = threadIdx.x, wave = tid >> 5, lane = tid & 31;
  const int half = lane >> 4, m = lane & 15;

  __shared__ __bf16 xT[SS][XC + 8];   // x chunk, transposed
  __shared__ float  tile[DD][SS];
  __shared__ float  rstd_s[SS];

  const float* wbase; int row0;
  if      (slot < 16) { wbase = wq; row0 = slot * DD; }
  else if (slot < 18) { wbase = wk; row0 = (slot - 16) * DD; }
  else                { wbase = wv; row0 = (slot - 18) * DD; }

  v8f acc = {};
  for (int kc = 0; kc < HID / XC; ++kc) {
    __syncthreads();
    // stage x[:, kc*XC .. +XC) transposed into LDS (coalesced global reads)
    for (int i = tid; i < XC * SS; i += 256) {
      int c = i >> 4, s = i & 15;
      xT[s][c] = (__bf16)x[(b * HID + kc * XC + c) * SS + s];
    }
    __syncthreads();
#pragma unroll 4
    for (int kk = 0; kk < XC / 32; ++kk) {
      const float* arow = wbase + (size_t)(row0 + wave * 16 + m) * HID + kc * XC + kk * 32;
      __builtin_prefetch(arow + 64, 0, 0);   // two K-steps ahead -> global_prefetch_b8
      V16 a; loadA_f32(a, arow, half);
      V16 bm;
      const u32x4* p = (const u32x4*)&xT[m][kk * 32 + 16 * half];
      bm.q[0] = p[0]; bm.q[1] = p[1];
      acc = wmma_bf16(a, bm, acc);
    }
  }
  __syncthreads();
#pragma unroll
  for (int r = 0; r < 8; ++r)
    tile[wave * 16 + 8 * half + r][m] = acc[r];
  __syncthreads();

  const int pos = *posp;
  if (slot < 18) {
    // RMSNorm stats per token column
    if (tid < SS) {
      float ss = 0.f;
      for (int d = 0; d < DD; ++d) { float v = tile[d][tid]; ss += v * v; }
      rstd_s[tid] = rsqrtf(ss * (1.0f / DD) + 1e-6f);
    }
    __syncthreads();
    const float* nw = (slot < 16) ? qnw : knw;
    for (int i = tid; i < (DD / 2) * SS; i += 256) {
      int j = i >> 4, s = i & 15;
      float r  = rstd_s[s];
      float x1 = tile[j][s]      * r * nw[j];
      float x2 = tile[j + 64][s] * r * nw[j + 64];
      float c  = cosb[s * 64 + j], sn = sinb[s * 64 + j];
      float y1 = x1 * c - x2 * sn;
      float y2 = x2 * c + x1 * sn;
      if (slot < 16) {
        __bf16* q = q_rot + ((size_t)(b * HH + slot) * SS + s) * DD;
        q[j] = (__bf16)y1; q[j + 64] = (__bf16)y2;
      } else {
        float* k = new_k + ((size_t)(b * KVH + (slot - 16)) * TT + pos + s) * DD;
        k[j] = y1; k[j + 64] = y2;
      }
    }
  } else {
    const int kvh = slot - 18;   // v: no norm, no rope
    for (int i = tid; i < DD * SS; i += 256) {
      int d = i >> 4, s = i & 15;
      new_v[((size_t)(b * KVH + kvh) * TT + pos + s) * DD + d] = tile[d][s];
    }
  }
}

// =====================================================================
// Kernel 2: flash-decoding attention. grid (NSPLIT, KVH, B), 8 waves = 8 GQA heads.
// K/V chunks are double-buffered in registers: stage -> barrier -> issue next
// chunk's global loads -> compute, so HBM latency overlaps WMMA/softmax work.
// =====================================================================
__global__ __launch_bounds__(256) void attn_kernel(
    const __bf16* __restrict__ q_rot, const float* __restrict__ kcache,
    const float* __restrict__ vcache, const float* __restrict__ mask,
    float* __restrict__ opart, float* __restrict__ mstat, float* __restrict__ lstat)
{
  const int split = blockIdx.x, kvh = blockIdx.y, b = blockIdx.z;
  const int tid = threadIdx.x, wave = tid >> 5, lane = tid & 31;
  const int half = lane >> 4, m = lane & 15;
  const int h = kvh * GG + wave;

  __shared__ __bf16 Klds[TCHUNK][DD + 8];     // row-major
  __shared__ __bf16 Vt[DD][TCHUNK + 8];       // transposed
  __shared__ __bf16 Plds[8][16][40];          // per-wave probs scratch

  // Q for this head: already bf16, direct b128 loads, zero conversion VALU.
  V16 aq[4];
#pragma unroll
  for (int kk = 0; kk < 4; ++kk) {
    const __bf16* qp = q_rot + ((size_t)(b * HH + h) * SS + m) * DD + kk * 32;
    aq[kk].q[0] = *(const u32x4*)(qp + 16 * half);
    aq[kk].q[1] = *(const u32x4*)(qp + 16 * half + 8);
  }

  const v8f vzero = {};
  v8f oacc[8];
#pragma unroll
  for (int i = 0; i < 8; ++i) oacc[i] = vzero;
  float rmax[8], rsum[8];
#pragma unroll
  for (int r = 0; r < 8; ++r) { rmax[r] = -1e30f; rsum[r] = 0.f; }

  const float scale = 0.08838834764831845f;  // 1/sqrt(128)
  const int tbase = split * (TT / NSPLIT);
  const int NCH = (TT / NSPLIT) / TCHUNK;    // 8 chunks per split
  const float* kptr0 = kcache + (size_t)(b * KVH + kvh) * TT * DD;
  const float* vptr0 = vcache + (size_t)(b * KVH + kvh) * TT * DD;

  f32x4 kreg[8], vreg[8];
  // prologue: fetch chunk 0
#pragma unroll
  for (int it = 0; it < 8; ++it) {
    int i = tid * 4 + it * 1024;
    int t = i >> 7, d = i & 127;
    kreg[it] = *(const f32x4*)(kptr0 + (size_t)(tbase + t) * DD + d);
    vreg[it] = *(const f32x4*)(vptr0 + (size_t)(tbase + t) * DD + d);
  }

  for (int ch = 0; ch < NCH; ++ch) {
    const int t0 = tbase + ch * TCHUNK;
    // stage current chunk (registers -> LDS, fp32 -> bf16)
#pragma unroll
    for (int it = 0; it < 8; ++it) {
      int i = tid * 4 + it * 1024;
      int t = i >> 7, d = i & 127;
      union { __bf16 h[4]; u32x2 u; } pk;
      pk.h[0] = (__bf16)kreg[it][0]; pk.h[1] = (__bf16)kreg[it][1];
      pk.h[2] = (__bf16)kreg[it][2]; pk.h[3] = (__bf16)kreg[it][3];
      *(u32x2*)&Klds[t][d] = pk.u;                       // one b64 store
      Vt[d + 0][t] = (__bf16)vreg[it][0];                // transposed scatter
      Vt[d + 1][t] = (__bf16)vreg[it][1];
      Vt[d + 2][t] = (__bf16)vreg[it][2];
      Vt[d + 3][t] = (__bf16)vreg[it][3];
    }
    __syncthreads();
    // issue next chunk's global loads now; they complete during compute below
    if (ch + 1 < NCH) {
      const int tn = t0 + TCHUNK;
#pragma unroll
      for (int it = 0; it < 8; ++it) {
        int i = tid * 4 + it * 1024;
        int t = i >> 7, d = i & 127;
        kreg[it] = *(const f32x4*)(kptr0 + (size_t)(tn + t) * DD + d);
        vreg[it] = *(const f32x4*)(vptr0 + (size_t)(tn + t) * DD + d);
      }
    }

#pragma unroll
    for (int pr = 0; pr < 2; ++pr) {            // two 16-wide score tiles per pass
      v8f sc0 = vzero, sc1 = vzero;
#pragma unroll
      for (int kk = 0; kk < 4; ++kk) {          // K = D = 128
        V16 bk0, bk1;
        const u32x4* p0 = (const u32x4*)&Klds[pr * 32 + m][kk * 32 + 16 * half];
        const u32x4* p1 = (const u32x4*)&Klds[pr * 32 + 16 + m][kk * 32 + 16 * half];
        bk0.q[0] = p0[0]; bk0.q[1] = p0[1];
        bk1.q[0] = p1[0]; bk1.q[1] = p1[1];
        sc0 = wmma_bf16(aq[kk], bk0, sc0);
        sc1 = wmma_bf16(aq[kk], bk1, sc1);
      }
      // online softmax over this 16x32 score block (rows live in 16-lane groups)
#pragma unroll
      for (int r = 0; r < 8; ++r) {
        const int srow = 8 * half + r;
        float v0 = sc0[r] * scale + mask[srow * TT + t0 + pr * 32 + m];
        float v1 = sc1[r] * scale + mask[srow * TT + t0 + pr * 32 + 16 + m];
        float pm = fmaxf(v0, v1);
#pragma unroll
        for (int off = 8; off >= 1; off >>= 1) pm = fmaxf(pm, __shfl_xor(pm, off, 16));
        float nm  = fmaxf(rmax[r], pm);
        float scf = __expf(rmax[r] - nm);
        rmax[r] = nm;
        float p0 = __expf(v0 - nm), p1 = __expf(v1 - nm);
        float rs = p0 + p1;
#pragma unroll
        for (int off = 8; off >= 1; off >>= 1) rs += __shfl_xor(rs, off, 16);
        rsum[r] = rsum[r] * scf + rs;
#pragma unroll
        for (int nt = 0; nt < 8; ++nt) oacc[nt][r] *= scf;
        Plds[wave][srow][m]      = (__bf16)p0;   // D-layout -> A-layout via LDS
        Plds[wave][srow][16 + m] = (__bf16)p1;
      }
      __syncthreads();
      // P(16x32) @ V(32x128): 8 N-tiles over D
      V16 ap;
      {
        const u32x4* pp0 = (const u32x4*)&Plds[wave][m][16 * half];
        const u32x4* pp1 = (const u32x4*)&Plds[wave][m][16 * half + 8];
        ap.q[0] = pp0[0]; ap.q[1] = pp1[0];
      }
#pragma unroll
      for (int nt = 0; nt < 8; ++nt) {
        V16 bv;
        const u32x4* pv = (const u32x4*)&Vt[nt * 16 + m][pr * 32 + 16 * half];
        bv.q[0] = pv[0]; bv.q[1] = pv[1];
        oacc[nt] = wmma_bf16(ap, bv, oacc[nt]);
      }
      __syncthreads();   // also gates the next chunk's LDS overwrite
    }
  }

  // partial output (unnormalized) + stats
  float* op = opart + ((size_t)(b * HH + h) * NSPLIT + split) * SS * DD;
#pragma unroll
  for (int nt = 0; nt < 8; ++nt)
#pragma unroll
    for (int r = 0; r < 8; ++r)
      op[(8 * half + r) * DD + nt * 16 + m] = oacc[nt][r];
  if (m == 0) {
    float* mp = mstat + ((size_t)(b * HH + h) * NSPLIT + split) * SS;
    float* lp = lstat + ((size_t)(b * HH + h) * NSPLIT + split) * SS;
#pragma unroll
    for (int r = 0; r < 8; ++r) { mp[8 * half + r] = rmax[r]; lp[8 * half + r] = rsum[r]; }
  }
}

// =====================================================================
// Kernel 3: combine flash-decoding partials. grid B*H*S blocks x 128 threads
// =====================================================================
__global__ __launch_bounds__(128) void combine_kernel(
    const float* __restrict__ opart, const float* __restrict__ mstat,
    const float* __restrict__ lstat, float* __restrict__ attn_out)
{
  const int row = blockIdx.x;            // (b*H + h)*S + s
  const int d = threadIdx.x;
  const int bh = row >> 4, s = row & 15;
  float M = -1e30f;
  for (int i = 0; i < NSPLIT; ++i)
    M = fmaxf(M, mstat[(bh * NSPLIT + i) * SS + s]);
  float O = 0.f, L = 0.f;
  for (int i = 0; i < NSPLIT; ++i) {
    float a = __expf(mstat[(bh * NSPLIT + i) * SS + s] - M);
    L += a * lstat[(bh * NSPLIT + i) * SS + s];
    O += a * opart[((size_t)(bh * NSPLIT + i) * SS + s) * DD + d];
  }
  attn_out[(size_t)row * DD + d] = O / fmaxf(L, 1e-20f);
}

// =====================================================================
// Kernel 4: output projection out = wo @ attn. grid (HID/128, B)
// =====================================================================
__global__ __launch_bounds__(256) void oproj_kernel(
    const float* __restrict__ attn, const float* __restrict__ wo,
    float* __restrict__ out)
{
  const int ob = blockIdx.x, b = blockIdx.y;
  const int tid = threadIdx.x, wave = tid >> 5, lane = tid & 31;
  const int half = lane >> 4, m = lane & 15;
  const int orow = ob * 128 + wave * 16;

  v8f acc = {};
#pragma unroll 4
  for (int kk = 0; kk < HID / 32; ++kk) {
    const float* arow = wo + (size_t)(orow + m) * HID + kk * 32;
    __builtin_prefetch(arow + 64, 0, 0);
    V16 a; loadA_f32(a, arow, half);
    // B operand: c = kk*32 + 16*half + e (16 contiguous c within one head's d-range)
    const int c0 = kk * 32 + 16 * half;
    const int hh = c0 >> 7, dl = c0 & 127;
    const float* bsrc = attn + ((size_t)(b * HH + hh) * SS + m) * DD + dl;
    V16 bm;
    f32x4 b0 = *(const f32x4*)(bsrc);
    f32x4 b1 = *(const f32x4*)(bsrc + 4);
    f32x4 b2 = *(const f32x4*)(bsrc + 8);
    f32x4 b3 = *(const f32x4*)(bsrc + 12);
#pragma unroll
    for (int e = 0; e < 4; ++e) {
      bm.h[e]      = (__bf16)b0[e];
      bm.h[4 + e]  = (__bf16)b1[e];
      bm.h[8 + e]  = (__bf16)b2[e];
      bm.h[12 + e] = (__bf16)b3[e];
    }
    acc = wmma_bf16(a, bm, acc);
  }
#pragma unroll
  for (int r = 0; r < 8; ++r)
    out[(size_t)(b * HID + orow + 8 * half + r) * SS + m] = acc[r];
}

// =====================================================================
extern "C" void kernel_launch(void* const* d_in, const int* in_sizes, int n_in,
                              void* d_out, int out_size, void* d_ws, size_t ws_size,
                              hipStream_t stream) {
  const float* x     = (const float*)d_in[0];
  const float* cosb  = (const float*)d_in[1];
  const float* sinb  = (const float*)d_in[2];
  const float* kv_k  = (const float*)d_in[3];
  const float* kv_v  = (const float*)d_in[4];
  const int*   posp  = (const int*)  d_in[5];
  const float* mask  = (const float*)d_in[6];
  const float* wq    = (const float*)d_in[7];
  const float* wk    = (const float*)d_in[8];
  const float* wv    = (const float*)d_in[9];
  const float* wo    = (const float*)d_in[10];
  const float* qnw   = (const float*)d_in[11];
  const float* knw   = (const float*)d_in[12];

  const size_t out_elems   = (size_t)BB * HID * SS;            // 262144
  const size_t cache_elems = (size_t)BB * KVH * TT * DD;       // 16777216

  float* out   = (float*)d_out;
  float* new_k = out + out_elems;
  float* new_v = new_k + cache_elems;

  float* ws      = (float*)d_ws;
  __bf16* q_rot  = (__bf16*)ws;                                // 262144 bf16 (uses 131072 floats)
  float* attn_o  = ws + 262144;                                // 262144
  float* opart   = ws + 524288;                                // B*H*NSPLIT*S*D = 4194304
  float* mstat   = ws + 524288 + 4194304;                      // 32768
  float* lstat   = mstat + 32768;                              // 32768

  // bulk cache pass-through (rows at `position` overwritten by qkv_kernel)
  hipMemcpyAsync(new_k, kv_k, cache_elems * sizeof(float), hipMemcpyDeviceToDevice, stream);
  hipMemcpyAsync(new_v, kv_v, cache_elems * sizeof(float), hipMemcpyDeviceToDevice, stream);

  qkv_kernel<<<dim3(HH + 2 * KVH, BB), 256, 0, stream>>>(
      x, wq, wk, wv, cosb, sinb, qnw, knw, posp, q_rot, new_k, new_v);

  attn_kernel<<<dim3(NSPLIT, KVH, BB), 256, 0, stream>>>(
      q_rot, new_k, new_v, mask, opart, mstat, lstat);

  combine_kernel<<<BB * HH * SS, 128, 0, stream>>>(opart, mstat, lstat, attn_o);

  oproj_kernel<<<dim3(HID / 128, BB), 256, 0, stream>>>(attn_o, wo, out);
}
